// ReadNet_20151986552975
// MI455X (gfx1250) — matrix operand order repaired and checked by reading, not compiled, not run on hardware
//
#include <hip/hip_runtime.h>
#include <math.h>

typedef __attribute__((ext_vector_type(16))) _Float16 v16h;
typedef __attribute__((ext_vector_type(8)))  _Float16 v8h;
typedef __attribute__((ext_vector_type(8)))  float    v8f;

#define D    256
#define HID  512
#define TPB  256
#define WPB  8     // waves per block (wave32)

// ---------------------------------------------------------------------------
// Kernel 1: one sweep over stm_emb (online softmax) + one sweep over ltm_emb.
// Per-block partials: (m, s, v[256]) for STM, l[256] for LTM.
// ---------------------------------------------------------------------------
__global__ __launch_bounds__(TPB) void k_partials(
    const float* __restrict__ x,
    const float* __restrict__ stm_emb, const float* __restrict__ stm_w,
    const float* __restrict__ ltm_emb, const float* __restrict__ ltm_w,
    int N, int nblocks,
    float* __restrict__ wsm, float* __restrict__ wss,
    float* __restrict__ wsv, float* __restrict__ wsl)
{
    __shared__ float sm[WPB];
    __shared__ float ss[WPB];
    __shared__ float sv[WPB][D];
    __shared__ float sl[WPB][D];

    const int tid   = threadIdx.x;
    const int wid   = tid >> 5;
    const int lane  = tid & 31;
    const int gwave = blockIdx.x * WPB + wid;
    const int nwav  = nblocks * WPB;

    // lane-private element slots within a 256-float row (two coalesced float4s)
    const int e0 = lane * 4;        // elements e0..e0+3
    const int e1 = 128 + lane * 4;  // elements e1..e1+3

    const float4 xa = *(const float4*)(x + e0);
    const float4 xb = *(const float4*)(x + e1);

    // ---- STM: online softmax, single pass ----
    float  m = -INFINITY, s = 0.f;
    float4 va = make_float4(0.f, 0.f, 0.f, 0.f);
    float4 vb = make_float4(0.f, 0.f, 0.f, 0.f);

    for (int row = gwave; row < N; row += nwav) {
        const float* rp = stm_emb + (size_t)row * D;
        const float4 ea = *(const float4*)(rp + e0);
        const float4 eb = *(const float4*)(rp + e1);
        // prefetch next row while we compute (speculative, no counter cost)
        if (row + nwav < N)
            __builtin_prefetch(stm_emb + (size_t)(row + nwav) * D + e0, 0, 0);

        float p = ea.x*xa.x + ea.y*xa.y + ea.z*xa.z + ea.w*xa.w
                + eb.x*xb.x + eb.y*xb.y + eb.z*xb.z + eb.w*xb.w;
        #pragma unroll
        for (int off = 16; off > 0; off >>= 1) p += __shfl_xor(p, off, 32);

        const float score = p * stm_w[row];
        const float mn    = fmaxf(m, score);
        const float corr  = __expf(m - mn);       // 0 when m == -inf
        const float e     = __expf(score - mn);
        s = s * corr + e;
        va.x = va.x*corr + e*ea.x;  va.y = va.y*corr + e*ea.y;
        va.z = va.z*corr + e*ea.z;  va.w = va.w*corr + e*ea.w;
        vb.x = vb.x*corr + e*eb.x;  vb.y = vb.y*corr + e*eb.y;
        vb.z = vb.z*corr + e*eb.z;  vb.w = vb.w*corr + e*eb.w;
        m = mn;
    }

    // ---- LTM: plain weighted sum, single pass ----
    float4 la = make_float4(0.f, 0.f, 0.f, 0.f);
    float4 lb = make_float4(0.f, 0.f, 0.f, 0.f);
    for (int row = gwave; row < N; row += nwav) {
        const float* rp = ltm_emb + (size_t)row * D;
        const float  w  = ltm_w[row];
        const float4 ea = *(const float4*)(rp + e0);
        const float4 eb = *(const float4*)(rp + e1);
        if (row + nwav < N)
            __builtin_prefetch(ltm_emb + (size_t)(row + nwav) * D + e0, 0, 0);
        la.x += w*ea.x; la.y += w*ea.y; la.z += w*ea.z; la.w += w*ea.w;
        lb.x += w*eb.x; lb.y += w*eb.y; lb.z += w*eb.z; lb.w += w*eb.w;
    }

    if (lane == 0) { sm[wid] = m; ss[wid] = s; }
    *(float4*)&sv[wid][e0] = va;  *(float4*)&sv[wid][e1] = vb;
    *(float4*)&sl[wid][e0] = la;  *(float4*)&sl[wid][e1] = lb;
    __syncthreads();

    // combine the 8 waves; thread d = tid handles one element
    float M = -INFINITY;
    #pragma unroll
    for (int w = 0; w < WPB; ++w) M = fmaxf(M, sm[w]);
    float S = 0.f, V = 0.f, L = 0.f;
    #pragma unroll
    for (int w = 0; w < WPB; ++w) {
        const float mw = sm[w];
        const float c  = (mw == -INFINITY) ? 0.f : __expf(mw - M);
        S += ss[w] * c;
        V += sv[w][tid] * c;
        L += sl[w][tid];
    }
    if (tid == 0) { wsm[blockIdx.x] = M; wss[blockIdx.x] = S; }
    wsv[(size_t)blockIdx.x * D + tid] = V;   // coalesced per block
    wsl[(size_t)blockIdx.x * D + tid] = L;
}

// ---------------------------------------------------------------------------
// Kernel 2: parallel reduction of per-block partials.
// Grid = D blocks; block d reduces over all B partials for element d.
// Partials are tiny (<=2MB) and L2-resident, so this stage is ~free.
// ---------------------------------------------------------------------------
__global__ __launch_bounds__(TPB) void k_combine(
    const float* __restrict__ x,
    const float* __restrict__ wsm, const float* __restrict__ wss,
    const float* __restrict__ wsv, const float* __restrict__ wsl,
    int nblocks, float* __restrict__ fused)
{
    __shared__ float rM[TPB];
    __shared__ float rS[TPB];
    __shared__ float rV[TPB];
    __shared__ float rL[TPB];

    const int d   = blockIdx.x;
    const int tid = threadIdx.x;

    // phase 1: global max of wsm
    float mml = -INFINITY;
    for (int b = tid; b < nblocks; b += TPB) mml = fmaxf(mml, wsm[b]);
    rM[tid] = mml;
    __syncthreads();
    #pragma unroll
    for (int sft = TPB/2; sft > 0; sft >>= 1) {
        if (tid < sft) rM[tid] = fmaxf(rM[tid], rM[tid + sft]);
        __syncthreads();
    }
    const float M = rM[0];
    __syncthreads();

    // phase 2: rescaled sums S, V, L
    float S = 0.f, V = 0.f, L = 0.f;
    for (int b = tid; b < nblocks; b += TPB) {
        const float mb = wsm[b];
        const float c  = (mb == -INFINITY) ? 0.f : __expf(mb - M);
        S += wss[b] * c;
        V += wsv[(size_t)b * D + d] * c;
        L += wsl[(size_t)b * D + d];
    }
    rS[tid] = S; rV[tid] = V; rL[tid] = L;
    __syncthreads();
    #pragma unroll
    for (int sft = TPB/2; sft > 0; sft >>= 1) {
        if (tid < sft) {
            rS[tid] += rS[tid + sft];
            rV[tid] += rV[tid + sft];
            rL[tid] += rL[tid + sft];
        }
        __syncthreads();
    }

    if (tid == 0) {
        fused[d]       = x[d];
        fused[D + d]   = rV[0] / rS[0];
        fused[2*D + d] = rL[0];
    }
}

// ---------------------------------------------------------------------------
// Kernel 3: MLP (1x768 @ 768x512 -> relu -> @ 512x256) on the WMMA pipe.
// M padded to 16. Trick: every lane loads ROW 0's data for its A fragment,
// so all 16 output rows are copies of the true result -- no zeroing, no
// predicated LDS loads; A-frag = two contiguous ds_load_b128 per k-step.
// ---------------------------------------------------------------------------
__global__ __launch_bounds__(TPB) void k_mlp(
    const float* __restrict__ fused,
    const float* __restrict__ w1, const float* __restrict__ b1,
    const float* __restrict__ w2, const float* __restrict__ b2,
    float* __restrict__ out)
{
    __shared__ _Float16 fh[3*D];   // fused, f16
    __shared__ _Float16 hh[HID];   // hidden activations, f16

    const int tid  = threadIdx.x;
    const int wid  = tid >> 5;
    const int lane = tid & 31;
    const int col  = lane & 15;          // B-matrix column within tile
    const int hi   = (lane >> 4) & 1;    // lane group (0..15 vs 16..31)

    for (int i = tid; i < 3*D; i += TPB) fh[i] = (_Float16)fused[i];
    __syncthreads();

    // ---- Layer 1: HID/16 = 32 N-tiles, 4 per wave ----
    for (int t = wid; t < HID/16; t += WPB) {
        v8f acc = {};
        for (int k0 = 0; k0 < 3*D; k0 += 32) {
            const int ab = k0 + (hi ? 8 : 0);
            const v8h a0 = *(const v8h*)&fh[ab];        // 16B-aligned LDS read
            const v8h a1 = *(const v8h*)&fh[ab + 16];
            v16h a;
            #pragma unroll
            for (int j = 0; j < 8; ++j) { a[j] = a0[j]; a[j + 8] = a1[j]; }

            v16h bf;
            const int n  = t*16 + col;
            const int kb = k0 + (hi ? 8 : 0);
            #pragma unroll
            for (int j = 0; j < 16; ++j)
                bf[j] = (_Float16)w1[(size_t)(kb + j + (j >= 8 ? 8 : 0)) * HID + n];

            acc = __builtin_amdgcn_wmma_f32_16x16x32_f16(
                false, a, false, bf, (short)0, acc, false, false);
        }
        // C row M=0 lives in VGPR0 of lanes 0..15 (N = lane); all rows equal.
        if (lane < 16) {
            const float hv = acc[0] + b1[t*16 + lane];
            hh[t*16 + lane] = (_Float16)fmaxf(hv, 0.f);
        }
    }
    __syncthreads();

    // ---- Layer 2: D/16 = 16 N-tiles, 2 per wave ----
    for (int t = wid; t < D/16; t += WPB) {
        v8f acc = {};
        for (int k0 = 0; k0 < HID; k0 += 32) {
            const int ab = k0 + (hi ? 8 : 0);
            const v8h a0 = *(const v8h*)&hh[ab];
            const v8h a1 = *(const v8h*)&hh[ab + 16];
            v16h a;
            #pragma unroll
            for (int j = 0; j < 8; ++j) { a[j] = a0[j]; a[j + 8] = a1[j]; }

            v16h bf;
            const int n  = t*16 + col;
            const int kb = k0 + (hi ? 8 : 0);
            #pragma unroll
            for (int j = 0; j < 16; ++j)
                bf[j] = (_Float16)w2[(size_t)(kb + j + (j >= 8 ? 8 : 0)) * D + n];

            acc = __builtin_amdgcn_wmma_f32_16x16x32_f16(
                false, a, false, bf, (short)0, acc, false, false);
        }
        if (lane < 16) out[t*16 + lane] = acc[0] + b2[t*16 + lane];
    }
}

// ---------------------------------------------------------------------------
extern "C" void kernel_launch(void* const* d_in, const int* in_sizes, int n_in,
                              void* d_out, int out_size, void* d_ws, size_t ws_size,
                              hipStream_t stream)
{
    const float* x       = (const float*)d_in[0];
    const float* stm_emb = (const float*)d_in[1];
    const float* stm_w   = (const float*)d_in[2];
    const float* ltm_emb = (const float*)d_in[3];
    const float* ltm_w   = (const float*)d_in[4];
    const float* w1      = (const float*)d_in[5];
    const float* b1      = (const float*)d_in[6];
    const float* w2      = (const float*)d_in[7];
    const float* b2      = (const float*)d_in[8];
    float* out = (float*)d_out;

    const int N = in_sizes[2];  // stm_w has N elements

    // Deterministic workspace budget: per block 2 + 2*D floats, plus fused[3*D].
    const size_t floats = ws_size / sizeof(float);
    int B = 1024;               // 8192 waves: fill the machine for the HBM sweep
    while (B > 1 && ((size_t)B * (2 + 2*D) + 3*D) > floats) B >>= 1;
    while (B > 1 && B * WPB > N) B >>= 1;

    float* wsm   = (float*)d_ws;
    float* wss   = wsm + B;
    float* wsv   = wss + B;
    float* wsl   = wsv + (size_t)B * D;
    float* fused = wsl + (size_t)B * D;

    k_partials<<<B, TPB, 0, stream>>>(x, stm_emb, stm_w, ltm_emb, ltm_w,
                                      N, B, wsm, wss, wsv, wsl);
    k_combine<<<D, TPB, 0, stream>>>(x, wsm, wss, wsv, wsl, B, fused);
    k_mlp<<<1, TPB, 0, stream>>>(fused, w1, b1, w2, b2, out);
}
